// FE_51711406243983
// MI455X (gfx1250) — matrix operand order repaired and checked by reading, not compiled, not run on hardware
//
#include <hip/hip_runtime.h>

// ---------------------------------------------------------------------------
// Types
// ---------------------------------------------------------------------------
typedef __bf16 bf16;
typedef __bf16 bf16x16 __attribute__((ext_vector_type(16)));
typedef float  f32x8   __attribute__((ext_vector_type(8)));
typedef unsigned int u32x4 __attribute__((ext_vector_type(4)));

#define TOK   50176      // B*H*W = 16*56*56
#define CCH   256
#define NWIN  1024       // 16 * 8 * 8 windows
#define HEADS 8
#define HD    32
#define WS7   7
#define NTOKW 49

__device__ inline f32x8 wmma_bf16(bf16x16 a, bf16x16 b, f32x8 c) {
  return __builtin_amdgcn_wmma_f32_16x16x32_bf16(false, a, false, b, (short)0, c,
                                                 false, false);
}

// A fragment: 16x32 (MxK) bf16. Lane m=lane&15 holds row M=m.
// half=0 (lanes 0-15):  K = kc+[0..7] then kc+[16..23]
// half=1 (lanes 16-31): K = kc+[8..15] then kc+[24..31]
__device__ inline bf16x16 load_frag_a(const bf16* base, int row0, int stride,
                                      int kc, int lane) {
  int m = lane & 15, half = (lane >> 4) & 1;
  const bf16* p = base + (size_t)(row0 + m) * stride + kc + half * 8;
  union { bf16x16 v; u32x4 u[2]; } f;
  f.u[0] = *(const u32x4*)(p);
  f.u[1] = *(const u32x4*)(p + 16);
  return f.v;
}

// B fragment from N-major (transposed) storage Bt[N][K]. Lane n=lane&15 holds
// column N=n; half=0 -> K=kc+[0..7],[16..23], half=1 -> K=kc+[8..15],[24..31].
__device__ inline bf16x16 load_frag_b(const bf16* base, int row0, int stride,
                                      int kc, int lane) {
  int n = lane & 15, half = (lane >> 4) & 1;
  const bf16* p = base + (size_t)(row0 + n) * stride + kc + half * 8;
  union { bf16x16 v; u32x4 u[2]; } f;
  f.u[0] = *(const u32x4*)(p);
  f.u[1] = *(const u32x4*)(p + 16);
  return f.v;
}

// CDNA5 async memory->LDS copy (ASYNCcnt tracked). Each active lane copies
// 16 bytes from its global address to its LDS offset. Generic pointers to
// __shared__ carry the LDS offset in addr[31:0] (flat LDS aperture).
__device__ inline void async_copy_b128(bf16* lds_dst, const bf16* gsrc) {
  unsigned loff = (unsigned)(unsigned long long)(void*)lds_dst;
  asm volatile("global_load_async_to_lds_b128 %0, %1, off"
               :
               : "v"(loff), "v"(gsrc)
               : "memory");
}
__device__ inline void wait_async0() {
  asm volatile("s_wait_asynccnt 0" ::: "memory");
}

__device__ inline float gelu_tanh(float x) {
  float x3 = x * x * x;
  float t = tanhf(0.7978845608028654f * (x + 0.044715f * x3));
  return 0.5f * x * (1.0f + t);
}

__device__ inline float wave_sum(float v) {
#pragma unroll
  for (int off = 16; off >= 1; off >>= 1) v += __shfl_xor(v, off, 32);
  return v;
}

// ---------------------------------------------------------------------------
// Weight prep: W[K][N] f32 -> Wt[N][K] bf16
// ---------------------------------------------------------------------------
__global__ void wtrans_kernel(const float* __restrict__ W, bf16* __restrict__ Wt,
                              int K, int N) {
  int i = blockIdx.x * 256 + threadIdx.x;
  if (i < K * N) {
    int k = i / N, n = i - k * N;
    Wt[(size_t)n * K + k] = (bf16)W[i];
  }
}

// deconv_w[ic][oc][3][3] f32 (flip kh,kw) -> Wc[oc][tap*256+ic] bf16
__global__ void convw_kernel(const float* __restrict__ W, bf16* __restrict__ Wt) {
  int i = blockIdx.x * 256 + threadIdx.x;
  if (i < 256 * 256 * 9) {
    int t = i % 9, rest = i / 9;
    int oc = rest % 256, ic = rest / 256;
    int u = t / 3, v = t - u * 3;
    float val = W[(((size_t)ic * 256 + oc) * 3 + (2 - u)) * 3 + (2 - v)];
    Wt[((size_t)oc * 9 + t) * 256 + ic] = (bf16)val;
  }
}

// ---------------------------------------------------------------------------
// LN1: x BCHW f32 -> XT (BHWC f32 residual stream) + BN (bf16 normalized)
// one wave per pixel, 8 waves per block
// ---------------------------------------------------------------------------
__global__ __launch_bounds__(256) void ln1_kernel(
    const float* __restrict__ x, const float* __restrict__ g,
    const float* __restrict__ be, float* __restrict__ xt, bf16* __restrict__ bn) {
  int lane = threadIdx.x & 31, wave = threadIdx.x >> 5;
  int pix = blockIdx.x * 8 + wave;
  int b = pix / 3136, rem = pix - b * 3136;
  const float* xp = x + (size_t)b * 256 * 3136 + rem;
  float v[8];
  float s = 0.f;
#pragma unroll
  for (int j = 0; j < 8; ++j) {
    v[j] = xp[(size_t)(lane + j * 32) * 3136];
    s += v[j];
  }
  s = wave_sum(s);
  float mean = s * (1.0f / 256.0f);
  float q = 0.f;
#pragma unroll
  for (int j = 0; j < 8; ++j) { float d = v[j] - mean; q += d * d; }
  q = wave_sum(q);
  float inv = rsqrtf(q * (1.0f / 256.0f) + 1e-5f);
  size_t ro = (size_t)pix * 256;
#pragma unroll
  for (int j = 0; j < 8; ++j) {
    int ch = lane + j * 32;
    xt[ro + ch] = v[j];
    bn[ro + ch] = (bf16)((v[j] - mean) * inv * g[ch] + be[ch]);
  }
}

// LN2: XT f32 -> BN bf16 (contiguous reads)
__global__ __launch_bounds__(256) void ln2_kernel(
    const float* __restrict__ xt, const float* __restrict__ g,
    const float* __restrict__ be, bf16* __restrict__ bn) {
  int lane = threadIdx.x & 31, wave = threadIdx.x >> 5;
  int pix = blockIdx.x * 8 + wave;
  size_t ro = (size_t)pix * 256;
  float v[8];
  float s = 0.f;
#pragma unroll
  for (int j = 0; j < 8; ++j) { v[j] = xt[ro + lane + j * 32]; s += v[j]; }
  s = wave_sum(s);
  float mean = s * (1.0f / 256.0f);
  float q = 0.f;
#pragma unroll
  for (int j = 0; j < 8; ++j) { float d = v[j] - mean; q += d * d; }
  q = wave_sum(q);
  float inv = rsqrtf(q * (1.0f / 256.0f) + 1e-5f);
#pragma unroll
  for (int j = 0; j < 8; ++j) {
    int ch = lane + j * 32;
    bn[ro + ch] = (bf16)((v[j] - mean) * inv * g[ch] + be[ch]);
  }
}

__global__ void tobf16_kernel(const float* __restrict__ in, bf16* __restrict__ out,
                              int n) {
  int i = blockIdx.x * 256 + threadIdx.x;
  if (i < n) out[i] = (bf16)in[i];
}

// ---------------------------------------------------------------------------
// Generic bf16 WMMA GEMM: C[M][N] = A[M][K] * Bt[N][K]^T + bias, with epilogue.
// Block: 256 threads (8 waves), tile 128x128; wave grid 2x4, wave tile 64x32.
// Double-buffered LDS staged with GLOBAL_LOAD_ASYNC_TO_LDS_B128 (ASYNCcnt).
// Requires M%128==0, N%128==0, K%32==0 (true for all call sites).
// ---------------------------------------------------------------------------
#define EPI_BF16      0   // outb = bf16(acc + bias)
#define EPI_GELU_BF16 1   // outb = bf16(gelu(acc + bias))
#define EPI_RESID_F32 2   // outf = acc + bias + resid   (in-place residual)

__global__ __launch_bounds__(256) void gemm_bf16_kernel(
    const bf16* __restrict__ A, const bf16* __restrict__ Bt,
    const float* __restrict__ bias, bf16* outb, float* outf, const float* resid,
    int M, int N, int K, int mode) {
  __shared__ __align__(16) bf16 As[2][128 * 32];
  __shared__ __align__(16) bf16 Bs[2][128 * 32];
  int nBlocks = N >> 7;
  int bm = blockIdx.x / nBlocks, bn = blockIdx.x - bm * nBlocks;
  int tid = threadIdx.x, lane = tid & 31, wave = tid >> 5;
  int waveM = wave >> 2, waveN = wave & 3;  // 2 x 4
  f32x8 acc[4][2] = {};

  // Per-thread staging chunks: 2 x 16B for A, 2 x 16B for B per k-step.
  const bf16* Abase = A + (size_t)bm * 128 * K;
  const bf16* Bbase = Bt + (size_t)bn * 128 * K;

  auto issue_async = [&](int buf, int k0) {
#pragma unroll
    for (int p = 0; p < 2; ++p) {
      int chunk = p * 256 + tid;          // 512 chunks of 8 elems
      int row = chunk >> 2;
      int col = (chunk & 3) * 8;
      async_copy_b128(&As[buf][row * 32 + col], &Abase[(size_t)row * K + k0 + col]);
      async_copy_b128(&Bs[buf][row * 32 + col], &Bbase[(size_t)row * K + k0 + col]);
    }
  };

  int nk = K >> 5;
  issue_async(0, 0);

  for (int ki = 0; ki < nk; ++ki) {
    int buf = ki & 1;
    wait_async0();        // this wave's async copies into As/Bs[buf] landed
    __syncthreads();      // every wave's copies landed
    if (ki + 1 < nk) issue_async(buf ^ 1, (ki + 1) * 32);

    bf16x16 bfr[2];
#pragma unroll
    for (int nt = 0; nt < 2; ++nt)
      bfr[nt] = load_frag_b(Bs[buf], waveN * 32 + nt * 16, 32, 0, lane);
#pragma unroll
    for (int mt = 0; mt < 4; ++mt) {
      bf16x16 afr = load_frag_a(As[buf], waveM * 64 + mt * 16, 32, 0, lane);
#pragma unroll
      for (int nt = 0; nt < 2; ++nt)
        acc[mt][nt] = wmma_bf16(afr, bfr[nt], acc[mt][nt]);
    }
    __syncthreads();      // reads of buf done before it is re-targeted
  }

  int nlane = lane & 15;
  int radd = (lane >> 4) * 8;
  int rowBase = bm * 128 + waveM * 64;
  int colBase = bn * 128 + waveN * 32;
#pragma unroll
  for (int mt = 0; mt < 4; ++mt)
#pragma unroll
    for (int nt = 0; nt < 2; ++nt) {
      int col = colBase + nt * 16 + nlane;
      float bv = bias[col];
#pragma unroll
      for (int r = 0; r < 8; ++r) {
        int row = rowBase + mt * 16 + r + radd;
        size_t idx = (size_t)row * N + col;
        float v = acc[mt][nt][r] + bv;
        if (mode == EPI_RESID_F32)       outf[idx] = v + resid[idx];
        else if (mode == EPI_GELU_BF16)  outb[idx] = (bf16)gelu_tanh(v);
        else                             outb[idx] = (bf16)v;
      }
    }
}

// ---------------------------------------------------------------------------
// Windowed attention: one (window, head) per 128-thread block.
// N=49 padded to 64, hd=32. Scores & P*V via WMMA; softmax in LDS.
// ---------------------------------------------------------------------------
__global__ __launch_bounds__(128) void attn_kernel(
    const bf16* __restrict__ qkv,      // [TOK][768] (q|k|v, head-major inside)
    const float* __restrict__ rel_bias,// [169][8]
    bf16* __restrict__ o) {            // [TOK][256]
  __shared__ __align__(16) bf16 Qs[64 * 32];
  __shared__ __align__(16) bf16 Ks[64 * 32];
  __shared__ __align__(16) bf16 Vt[32 * 64];
  __shared__ __align__(16) bf16 Pb[64 * 64];
  __shared__ float Sc[64 * 64];

  int tid = threadIdx.x, lane = tid & 31, wave = tid >> 5;
  int task = blockIdx.x;                 // 8192 tasks
  int head = task & 7, win = task >> 3;
  int b = win >> 6, w64 = win & 63;
  int wy = w64 >> 3, wx = w64 & 7;

  // token -> flat BHWC row
  auto tokrow = [&](int n) {
    int ty = n / 7, tx = n - ty * 7;
    int y = wy * 7 + ty, x = wx * 7 + tx;
    return (b * 56 + y) * 56 + x;
  };

  // ---- stage Q, K (64x32, rows >=49 zero) ----
  int lrow = tid >> 2;              // 0..31
  int lcol = (tid & 3) * 8;         // 0,8,16,24
#pragma unroll
  for (int it = 0; it < 2; ++it) {
    int n = it * 32 + lrow;
    u32x4 qv = {0, 0, 0, 0}, kv = {0, 0, 0, 0};
    if (n < NTOKW) {
      size_t base = (size_t)tokrow(n) * 768;
      qv = *(const u32x4*)&qkv[base + head * 32 + lcol];
      kv = *(const u32x4*)&qkv[base + 256 + head * 32 + lcol];
    }
    *(u32x4*)&Qs[n * 32 + lcol] = qv;
    *(u32x4*)&Ks[n * 32 + lcol] = kv;
  }
  // ---- stage V transposed: Vt[d][m] ----
  {
    int d = tid & 31, mseg = tid >> 5;  // 4 segments of 16 m
    for (int m = mseg * 16; m < mseg * 16 + 16; ++m) {
      bf16 v = (bf16)0.0f;
      if (m < NTOKW) v = qkv[(size_t)tokrow(m) * 768 + 512 + head * 32 + d];
      Vt[d * 64 + m] = v;
    }
  }
  __syncthreads();

  // ---- scores: 4x4 WMMA tiles; wave w handles row-tile mt=w ----
  {
    int mt = wave;
    bf16x16 afr = load_frag_a(Qs, mt * 16, 32, 0, lane);
#pragma unroll
    for (int nt = 0; nt < 4; ++nt) {
      bf16x16 bfr = load_frag_b(Ks, nt * 16, 32, 0, lane);
      f32x8 c = {};
      c = wmma_bf16(afr, bfr, c);
      int coln = nt * 16 + (lane & 15);
      int radd = (lane >> 4) * 8;
#pragma unroll
      for (int r = 0; r < 8; ++r) Sc[(mt * 16 + r + radd) * 64 + coln] = c[r];
    }
  }
  __syncthreads();

  // ---- softmax over m (49 valid) with scale + relative position bias ----
  const float scale = 0.17677669529663687f;  // 1/sqrt(32)
  {
    int n = tid;
    if (n < 64) {
      if (n < NTOKW) {
        int ny = n / 7, nx = n - ny * 7;
        float mx = -1e30f;
        for (int m = 0; m < NTOKW; ++m) {
          int my = m / 7, mc = m - my * 7;
          float bb = rel_bias[((ny - my + 6) * 13 + (nx - mc + 6)) * 8 + head];
          float s = Sc[n * 64 + m] * scale + bb;
          mx = fmaxf(mx, s);
        }
        float sum = 0.f;
        for (int m = 0; m < NTOKW; ++m) {
          int my = m / 7, mc = m - my * 7;
          float bb = rel_bias[((ny - my + 6) * 13 + (nx - mc + 6)) * 8 + head];
          float e = __expf(Sc[n * 64 + m] * scale + bb - mx);
          sum += e;
          Pb[n * 64 + m] = (bf16)e;
        }
        float inv = 1.0f / sum;
        for (int m = 0; m < NTOKW; ++m)
          Pb[n * 64 + m] = (bf16)((float)Pb[n * 64 + m] * inv);
        for (int m = NTOKW; m < 64; ++m) Pb[n * 64 + m] = (bf16)0.0f;
      } else {
        for (int m = 0; m < 64; ++m) Pb[n * 64 + m] = (bf16)0.0f;
      }
    }
  }
  __syncthreads();

  // ---- O = P (64x64) x V (64x32): wave w handles row-tile mt=w, nt 0..1 ----
  {
    int mt = wave;
#pragma unroll
    for (int nt = 0; nt < 2; ++nt) {
      f32x8 c = {};
#pragma unroll
      for (int ks = 0; ks < 2; ++ks) {
        bf16x16 afr = load_frag_a(Pb, mt * 16, 64, ks * 32, lane);
        bf16x16 bfr = load_frag_b(Vt, nt * 16, 64, ks * 32, lane);
        c = wmma_bf16(afr, bfr, c);
      }
      int dcol = nt * 16 + (lane & 15);
      int radd = (lane >> 4) * 8;
#pragma unroll
      for (int r = 0; r < 8; ++r) {
        int n = mt * 16 + r + radd;
        if (n < NTOKW)
          o[(size_t)tokrow(n) * 256 + head * 32 + dcol] = (bf16)c[r];
      }
    }
  }
}

// ---------------------------------------------------------------------------
// 3x3 conv (ConvTranspose2d s1p1 == conv with flipped kernel) as implicit GEMM.
// Block: 128 threads (4 waves). Tile: 64 pixels (8x8 spatial) x 128 oc.
// Wave grid 2x2; wave tile 32 rows x 64 cols. K loop: 9 taps x 8 chunks of 32.
// Boundary-masked A loads stay synchronous (need zero fill); weight tile gets
// a prefetch one k-chunk ahead.
// ---------------------------------------------------------------------------
__global__ __launch_bounds__(128) void conv_kernel(
    const bf16* __restrict__ xb,   // [B][56][56][256]
    const bf16* __restrict__ wt,   // [256][2304]  (oc x (tap*256+ic))
    const float* __restrict__ bias,
    float* __restrict__ out) {     // [B][256][56][56]
  __shared__ __align__(16) bf16 As[64 * 32];
  __shared__ __align__(16) bf16 Bs[128 * 32];
  int bid = blockIdx.x;
  int bn = bid & 1; bid >>= 1;
  int bxs = bid % 7; bid /= 7;
  int bys = bid % 7; int b = bid / 7;
  int tid = threadIdx.x, lane = tid & 31, wave = tid >> 5;
  int waveM = wave >> 1, waveN = wave & 1;
  f32x8 acc[2][4] = {};

  for (int tap = 0; tap < 9; ++tap) {
    int u = tap / 3, v = tap - u * 3;
    for (int kc = 0; kc < 256; kc += 32) {
      // A: 64x32 (pixels x ic), zero-padded at image borders
#pragma unroll
      for (int p = 0; p < 2; ++p) {
        int chunk = p * 128 + tid;       // 256 chunks of 8
        int prow = chunk >> 2;
        int pcol = (chunk & 3) * 8;
        int py = prow >> 3, px = prow & 7;
        int y = bys * 8 + py + u - 1;
        int x = bxs * 8 + px + v - 1;
        u32x4 val = {0, 0, 0, 0};
        if (y >= 0 && y < 56 && x >= 0 && x < 56)
          val = *(const u32x4*)&xb[(((size_t)b * 56 + y) * 56 + x) * 256 + kc + pcol];
        *(u32x4*)&As[prow * 32 + pcol] = val;
      }
      // B: 128x32 weight tile (+ prefetch next chunk)
#pragma unroll
      for (int p = 0; p < 4; ++p) {
        int chunk = p * 128 + tid;       // 512 chunks of 8
        int prow = chunk >> 2;
        int pcol = (chunk & 3) * 8;
        *(u32x4*)&Bs[prow * 32 + pcol] =
            *(const u32x4*)&wt[(size_t)(bn * 128 + prow) * 2304 + tap * 256 + kc + pcol];
      }
      if (kc + 32 < 256)
        __builtin_prefetch(&wt[(size_t)(bn * 128 + (tid >> 2)) * 2304 + tap * 256 + kc + 32], 0, 3);
      __syncthreads();
      bf16x16 bfr[4];
#pragma unroll
      for (int nt = 0; nt < 4; ++nt)
        bfr[nt] = load_frag_b(Bs, waveN * 64 + nt * 16, 32, 0, lane);
#pragma unroll
      for (int mt = 0; mt < 2; ++mt) {
        bf16x16 afr = load_frag_a(As, waveM * 32 + mt * 16, 32, 0, lane);
#pragma unroll
        for (int nt = 0; nt < 4; ++nt)
          acc[mt][nt] = wmma_bf16(afr, bfr[nt], acc[mt][nt]);
      }
      __syncthreads();
    }
  }

  int nlane = lane & 15, radd = (lane >> 4) * 8;
#pragma unroll
  for (int mt = 0; mt < 2; ++mt)
#pragma unroll
    for (int nt = 0; nt < 4; ++nt) {
      int oc = bn * 128 + waveN * 64 + nt * 16 + nlane;
      float bv = bias[oc];
#pragma unroll
      for (int r = 0; r < 8; ++r) {
        int prow = waveM * 32 + mt * 16 + r + radd;   // 0..63
        int py = prow >> 3, px = prow & 7;
        int y = bys * 8 + py, x = bxs * 8 + px;
        out[(((size_t)b * 256 + oc) * 56 + y) * 56 + x] = acc[mt][nt][r] + bv;
      }
    }
}

// ---------------------------------------------------------------------------
// Host launcher
// ---------------------------------------------------------------------------
extern "C" void kernel_launch(void* const* d_in, const int* in_sizes, int n_in,
                              void* d_out, int out_size, void* d_ws, size_t ws_size,
                              hipStream_t stream) {
  const float* x      = (const float*)d_in[0];
  const float* ln1_g  = (const float*)d_in[1];
  const float* ln1_b  = (const float*)d_in[2];
  const float* qkv_w  = (const float*)d_in[3];
  const float* qkv_b  = (const float*)d_in[4];
  const float* proj_w = (const float*)d_in[5];
  const float* proj_b = (const float*)d_in[6];
  const float* rel_b  = (const float*)d_in[7];
  const float* ln2_g  = (const float*)d_in[8];
  const float* ln2_b  = (const float*)d_in[9];
  const float* mlp_w1 = (const float*)d_in[10];
  const float* mlp_b1 = (const float*)d_in[11];
  const float* mlp_w2 = (const float*)d_in[12];
  const float* mlp_b2 = (const float*)d_in[13];
  const float* dc_w   = (const float*)d_in[14];
  const float* dc_b   = (const float*)d_in[15];
  float* out = (float*)d_out;

  char* ws = (char*)d_ws;
  size_t off = 0;
  auto alloc = [&](size_t bytes) -> char* {
    char* p = ws + off;
    off += (bytes + 255) & ~(size_t)255;
    return p;
  };
  float* XT    = (float*)alloc((size_t)TOK * 256 * 4);
  bf16*  BN    = (bf16*) alloc((size_t)TOK * 256 * 2);
  bf16*  QKV   = (bf16*) alloc((size_t)TOK * 768 * 2);
  bf16*  O     = (bf16*) alloc((size_t)TOK * 256 * 2);
  bf16*  MID   = (bf16*) alloc((size_t)TOK * 1024 * 2);
  bf16*  XB    = (bf16*) alloc((size_t)TOK * 256 * 2);
  bf16*  WQKV  = (bf16*) alloc((size_t)768 * 256 * 2);
  bf16*  WPROJ = (bf16*) alloc((size_t)256 * 256 * 2);
  bf16*  WM1   = (bf16*) alloc((size_t)1024 * 256 * 2);
  bf16*  WM2   = (bf16*) alloc((size_t)256 * 1024 * 2);
  bf16*  WC    = (bf16*) alloc((size_t)256 * 2304 * 2);

  // 1) weights -> bf16 transposed
  wtrans_kernel<<<(256 * 768 + 255) / 256, 256, 0, stream>>>(qkv_w, WQKV, 256, 768);
  wtrans_kernel<<<(256 * 256 + 255) / 256, 256, 0, stream>>>(proj_w, WPROJ, 256, 256);
  wtrans_kernel<<<(256 * 1024 + 255) / 256, 256, 0, stream>>>(mlp_w1, WM1, 256, 1024);
  wtrans_kernel<<<(1024 * 256 + 255) / 256, 256, 0, stream>>>(mlp_w2, WM2, 1024, 256);
  convw_kernel<<<(256 * 256 * 9 + 255) / 256, 256, 0, stream>>>(dc_w, WC);

  // 2) LN1: BCHW -> XT (f32) + BN (bf16)
  ln1_kernel<<<TOK / 8, 256, 0, stream>>>(x, ln1_g, ln1_b, XT, BN);

  // 3) QKV GEMM: [TOK,256] x [256,768] -> QKV bf16
  gemm_bf16_kernel<<<(TOK / 128) * (768 / 128), 256, 0, stream>>>(
      BN, WQKV, qkv_b, QKV, nullptr, nullptr, TOK, 768, 256, EPI_BF16);

  // 4) attention -> O bf16
  attn_kernel<<<NWIN * HEADS, 128, 0, stream>>>(QKV, rel_b, O);

  // 5) proj GEMM + residual (in-place into XT)
  gemm_bf16_kernel<<<(TOK / 128) * (256 / 128), 256, 0, stream>>>(
      O, WPROJ, proj_b, nullptr, XT, XT, TOK, 256, 256, EPI_RESID_F32);

  // 6) LN2: XT -> BN
  ln2_kernel<<<TOK / 8, 256, 0, stream>>>(XT, ln2_g, ln2_b, BN);

  // 7) MLP1 GEMM + GELU -> MID bf16
  gemm_bf16_kernel<<<(TOK / 128) * (1024 / 128), 256, 0, stream>>>(
      BN, WM1, mlp_b1, MID, nullptr, nullptr, TOK, 1024, 256, EPI_GELU_BF16);

  // 8) MLP2 GEMM + residual -> XT
  gemm_bf16_kernel<<<(TOK / 128) * (256 / 128), 256, 0, stream>>>(
      MID, WM2, mlp_b2, nullptr, XT, XT, TOK, 256, 1024, EPI_RESID_F32);

  // 9) XT -> bf16 for conv
  tobf16_kernel<<<(TOK * 256 + 255) / 256, 256, 0, stream>>>(XT, XB, TOK * 256);

  // 10) 3x3 conv (implicit GEMM) -> d_out BCHW f32 (+bias)
  conv_kernel<<<16 * 7 * 7 * 2, 128, 0, stream>>>(XB, WC, dc_b, out);
}
